// GFlowLoss_41386304864874
// MI455X (gfx1250) — compile-verified, matrix operand order checked by reading, not run on hardware
//
#include <hip/hip_runtime.h>

typedef _Float16 v16h __attribute__((ext_vector_type(16)));
typedef float    v8f  __attribute__((ext_vector_type(8)));

#define ADAM_B1  0.9f
#define ADAM_B2  0.999f
#define ADAM_EPS 1e-8f
#define LRATE    0.1f
#define CEPS     0.1f
#define NITER    200
#define COL_INF  1e30f

// Dynamic LDS layout (bytes):
//   [0, 256K)      : packed bf16 (m | v) Adam state, one u32 per matrix element (256x256)
//   [256K, +16K)   : f16 per-wave column partials colH[32][256] (rows 16..31 stay zero)
//   then           : sT[256], sR[256], sC[256], sWP[32]  (f32)
#define SMEM_MV    (65536u * 4u)
#define SMEM_COLH  (32u * 256u * 2u)
#define SMEM_BYTES (SMEM_MV + SMEM_COLH + (256u * 3u + 32u) * 4u)

__device__ __forceinline__ float    bf_lo(unsigned u) { return __uint_as_float(u << 16); }
__device__ __forceinline__ float    bf_hi(unsigned u) { return __uint_as_float(u & 0xFFFF0000u); }
__device__ __forceinline__ unsigned bf_pack(float lo, float hi) {
  return (__float_as_uint(lo) >> 16) | (__float_as_uint(hi) & 0xFFFF0000u);
}
// sigmoid via hardware exp2/rcp
__device__ __forceinline__ float fast_sigmoid(float x) {
  return __builtin_amdgcn_rcpf(1.f + __builtin_amdgcn_exp2f(-x * 1.44269504f));
}

__global__ __launch_bounds__(512, 1)
void gflow_main(const float* __restrict__ adj,     // [128,256,256]
                const float* __restrict__ gl0,     // [128,248,248]
                const float* __restrict__ T0,      // [128,256]
                float* __restrict__ ws)            // [128*2] partials
{
  extern __shared__ char smem[];
  unsigned*  mv   = (unsigned*)smem;                       // 65536 words
  _Float16*  colH = (_Float16*)(smem + SMEM_MV);           // [32][256]
  float*     sT   = (float*)(smem + SMEM_MV + SMEM_COLH);  // [256]
  float*     sR   = sT + 256;                              // [256]
  float*     sC   = sR + 256;                              // [256]
  float*     sWP  = sC + 256;                              // [32]

  const int b    = blockIdx.x;
  const int tid  = threadIdx.x;
  const int w    = tid >> 5;        // wave 0..15, owns rows 16w..16w+15
  const int l    = tid & 31;        // lane, owns cols l+32k
  const int hl   = l >> 4;          // WMMA half select
  const int n16  = l & 15;

  const float kp = 1.f / (128.f * 248.f);           // parity mean scale
  const float kc = 1.f / (128.f * 248.f * 248.f);   // causal mean scale
  const float ks = 0.1f * kc;                       // sparsity grad scale

  const float* adjb = adj + (size_t)b * 65536;
  const float* glb  = gl0 + (size_t)b * 61504;

  // ---- init: G_logits into regs, zero Adam state, zero colH rows 16..31 ----
  float gl[128];
#pragma unroll
  for (int r = 0; r < 16; ++r) {
    const int i = 16 * w + r;
#pragma unroll
    for (int k = 0; k < 8; ++k) {
      const int j = l + 32 * k;
      gl[r * 8 + k] = (i < 248 && j < 248) ? glb[i * 248 + j] : 0.f;
      mv[i * 256 + j] = 0u;
    }
  }
#pragma unroll
  for (int k = 0; k < 8; ++k) colH[(16 + w) * 256 + l + 32 * k] = (_Float16)0.f;

  float Tval = 0.f, mT = 0.f, vT = 0.f;
  if (tid < 256) { Tval = T0[b * 256 + tid]; sT[tid] = Tval; }
  __syncthreads();

  // ================= 200 fused Adam steps =================
  float pb1 = 1.f, pb2 = 1.f;
  for (int t = 1; t <= NITER; ++t) {
    pb1 *= ADAM_B1; pb2 *= ADAM_B2;
    const float ibc2 = __builtin_amdgcn_rcpf(1.f - pb2);
    const float lrm  = LRATE * __builtin_amdgcn_rcpf(1.f - pb1);

    // column-T values; padded columns get +INF so pen==0 kills every term
    float tv[8];
#pragma unroll
    for (int k = 0; k < 8; ++k) {
      const int j = l + 32 * k;
      tv[k] = (j < 248) ? sT[8 + j] : COL_INF;
    }
    float colacc[8];
#pragma unroll
    for (int k = 0; k < 8; ++k) colacc[k] = 0.f;

#pragma unroll
    for (int r = 0; r < 16; ++r) {
      const int i = 16 * w + r;
      if (i < 248) {                          // wave-uniform (only wave 15 trims)
        const float* arow = adjb + i * 256 + 8 + l;
        const float tue = sT[i] + CEPS;       // wave-uniform broadcast read

        float a8[8], g8[8];
#pragma unroll
        for (int k = 0; k < 7; ++k) a8[k] = arow[32 * k];
        a8[7] = (l < 24) ? arow[224] : 0.f;   // only k=7 can run past col 247

        float sAp = 0.f, Rp = 0.f;
#pragma unroll
        for (int k = 0; k < 8; ++k) {
          const float g = fast_sigmoid(gl[r * 8 + k]);
          g8[k] = g;
          float pen = tue - tv[k];
          const float Mg = (pen > 0.f) ? g : 0.f;
          sAp += a8[k] * g;
          Rp  += Mg;
          colacc[k] += Mg;
        }
#pragma unroll
        for (int off = 16; off > 0; off >>= 1) {  // wave32 butterfly
          sAp += __shfl_xor(sAp, off, 32);
          Rp  += __shfl_xor(Rp,  off, 32);
        }
        if (l == r) sR[i] = Rp;
        // parity dL/dS (hardware sin/cos in "turns": sin(pi*S/2)=sin_hw(S/4))
        const float q = sAp * 0.25f;
        const float s = __builtin_amdgcn_sinf(q);
        const float c = __builtin_amdgcn_cosf(q);
        const float dS = kp * 6.28318531f * (s * s - 1.f) * s * c;

        // grad + Adam for this row (m|v packed bf16 in LDS)
#pragma unroll
        for (int k = 0; k < 8; ++k) {
          const float g = g8[k];
          float pen = tue - tv[k];
          pen = pen > 0.f ? pen : 0.f;
          const float sgn = (g >= 0.5f) ? ks : -ks;
          float grad = dS * a8[k] + kc * pen + sgn;
          grad *= g * (1.f - g);

          const int idx = i * 256 + l + 32 * k;
          const unsigned pk = mv[idx];
          float m_ = bf_lo(pk), v_ = bf_hi(pk);
          m_ = ADAM_B1 * m_ + (1.f - ADAM_B1) * grad;
          v_ = ADAM_B2 * v_ + (1.f - ADAM_B2) * grad * grad;
          const float upd = lrm * m_ *
              __builtin_amdgcn_rcpf(__builtin_amdgcn_sqrtf(v_ * ibc2) + ADAM_EPS);
          gl[r * 8 + k] -= upd;
          mv[idx] = bf_pack(m_, v_);
        }
        __builtin_amdgcn_sched_barrier(0);    // cap live range at one row
      }
    }
#pragma unroll
    for (int k = 0; k < 8; ++k) colH[w * 256 + l + 32 * k] = (_Float16)colacc[k];
    __syncthreads();

    // ---- cross-wave column sums: ones(16x32) x colH-tile(32x16) via WMMA ----
    // K=32 collapses all 16 wave partial rows (+16 zero rows) in one
    // v_wmma_f32_16x16x32_f16 per 16-column chunk; all 16 waves participate.
    {
      const int n0 = w * 16;
      v16h va, vb;
#pragma unroll
      for (int m2 = 0; m2 < 16; ++m2) va[m2] = (_Float16)1.0f;
#pragma unroll
      for (int v2 = 0; v2 < 8; ++v2) {     // B: VGPR v2 holds rows {2v2,2v2+1}+16*half
        vb[2 * v2]     = colH[(2 * v2     + 16 * hl) * 256 + n0 + n16];
        vb[2 * v2 + 1] = colH[(2 * v2 + 1 + 16 * hl) * 256 + n0 + n16];
      }
      v8f acc = {};
      acc = __builtin_amdgcn_wmma_f32_16x16x32_f16(
          false, va, false, vb, (short)0, acc, false, false);
      if (l < 16) sC[n0 + l] = acc[0];     // D row 0, lanes 0..15 = N 0..15
    }
    __syncthreads();

    // ---- Adam on T (per-node state in regs of threads 0..255) ----
    if (tid < 256) {
      const float Rn = (tid < 248) ? sR[tid]     : 0.f;   // V_MEAS = 0..247
      const float Cn = (tid >= 8)  ? sC[tid - 8] : 0.f;   // V_CORR = 8..255
      const float gT = kc * (Rn - Cn);
      mT = ADAM_B1 * mT + (1.f - ADAM_B1) * gT;
      vT = ADAM_B2 * vT + (1.f - ADAM_B2) * gT * gT;
      Tval -= lrm * mT *
          __builtin_amdgcn_rcpf(__builtin_amdgcn_sqrtf(vT * ibc2) + ADAM_EPS);
      sT[tid] = Tval;
    }
    __syncthreads();
  }

  // ================= final losses (with final G, final T) =================
  {
    float tv[8];
#pragma unroll
    for (int k = 0; k < 8; ++k) {
      const int j = l + 32 * k;
      tv[k] = (j < 248) ? sT[8 + j] : COL_INF;
    }
    float ppart = 0.f, cpart = 0.f;
#pragma unroll
    for (int r = 0; r < 16; ++r) {
      const int i = 16 * w + r;
      if (i < 248) {
        const float* arow = adjb + i * 256 + 8 + l;
        const float tue = sT[i] + CEPS;
        float a8[8];
#pragma unroll
        for (int k = 0; k < 7; ++k) a8[k] = arow[32 * k];
        a8[7] = (l < 24) ? arow[224] : 0.f;

        float sAp = 0.f, cp = 0.f;
#pragma unroll
        for (int k = 0; k < 8; ++k) {
          const float g = fast_sigmoid(gl[r * 8 + k]);
          float pen = tue - tv[k];
          pen = pen > 0.f ? pen : 0.f;
          sAp += a8[k] * g;
          cp  += g * pen;
        }
#pragma unroll
        for (int off = 16; off > 0; off >>= 1) {
          sAp += __shfl_xor(sAp, off, 32);
          cp  += __shfl_xor(cp,  off, 32);
        }
        if (l == 0) {
          const float s = __builtin_amdgcn_sinf(sAp * 0.25f);
          const float d = s * s - 1.f;
          ppart += d * d;
          cpart += cp;
        }
        __builtin_amdgcn_sched_barrier(0);
      }
    }
    if (l == 0) { sWP[w] = ppart; sWP[16 + w] = cpart; }
  }
  __syncthreads();
  if (tid == 0) {
    float pp = 0.f, cc = 0.f;
    for (int i = 0; i < 16; ++i) { pp += sWP[i]; cc += sWP[16 + i]; }
    ws[2 * b] = pp; ws[2 * b + 1] = cc;     // deterministic per-batch partials
  }
}

__global__ void gflow_finalize(const float* __restrict__ ws, float* __restrict__ out) {
  if (blockIdx.x == 0 && threadIdx.x == 0) {
    float pp = 0.f, cc = 0.f;
    for (int b = 0; b < 128; ++b) { pp += ws[2 * b]; cc += ws[2 * b + 1]; }
    pp *= 1.f / (128.f * 248.f);
    cc *= 1.f / (128.f * 248.f * 248.f);
    out[0] = pp + cc; out[1] = pp; out[2] = cc;
  }
}

extern "C" void kernel_launch(void* const* d_in, const int* in_sizes, int n_in,
                              void* d_out, int out_size, void* d_ws, size_t ws_size,
                              hipStream_t stream) {
  (void)in_sizes; (void)n_in; (void)out_size; (void)ws_size;
  const float* adj = (const float*)d_in[0];   // [128,256,256]
  const float* gl0 = (const float*)d_in[1];   // [128,248,248]
  const float* T0  = (const float*)d_in[2];   // [128,256]
  float* ws  = (float*)d_ws;
  float* out = (float*)d_out;

  // >64KB dynamic LDS: opt in (idempotent, capture-safe; ignore return value)
  (void)hipFuncSetAttribute((const void*)gflow_main,
                            hipFuncAttributeMaxDynamicSharedMemorySize,
                            (int)SMEM_BYTES);

  gflow_main<<<128, 512, SMEM_BYTES, stream>>>(adj, gl0, T0, ws);
  gflow_finalize<<<1, 32, 0, stream>>>(ws, out);
}
